// MPN_75754633167331
// MI455X (gfx1250) — matrix operand order repaired and checked by reading, not compiled, not run on hardware
//
#include <hip/hip_runtime.h>
#include <hip/hip_bf16.h>

// ---------------- constants from the reference ----------------
#define Hc     128       // hidden channels
#define NRc    6         // num_radial
#define NSRc   42        // NS*NR
#define NBc    8         // num_bilinear
#define OUTc   128
#define NLAYc  3
#define NNODES 20000

typedef __attribute__((ext_vector_type(16))) __bf16 v16bf;
typedef __attribute__((ext_vector_type(8)))  float  v8f;
typedef __attribute__((__vector_size__(4 * sizeof(int)))) int v4i;

// ---- CDNA5 async global->LDS copy (ASYNCcnt-tracked), guarded fallback ----
#if __has_builtin(__builtin_amdgcn_global_load_async_to_lds_b128) && \
    __has_builtin(__builtin_amdgcn_s_wait_asynccnt)
#define HAVE_ASYNC_LDS 1
#endif

__device__ __forceinline__ void async_copy_b128(const __bf16* gsrc, __bf16* ldst) {
#ifdef HAVE_ASYNC_LDS
    __builtin_amdgcn_global_load_async_to_lds_b128(
        (__attribute__((address_space(1))) v4i*)gsrc,
        (__attribute__((address_space(3))) v4i*)ldst, 0, 0);
#else
    *(uint4*)ldst = *(const uint4*)gsrc;
#endif
}

__device__ __forceinline__ void async_wait() {
#ifdef HAVE_ASYNC_LDS
    __builtin_amdgcn_s_wait_asynccnt(0);
#endif
}

// 32KB bf16 weight: global -> LDS in b128 chunks (async if available)
__device__ __forceinline__ void stage_w(__bf16* dst, const __bf16* src, int tid) {
#pragma unroll 4
    for (int k = tid; k < Hc * Hc / 8; k += 128)
        async_copy_b128(src + (size_t)k * 8, dst + (size_t)k * 8);
}

// fast silu: v * rcp(1+exp(-v))
__device__ __forceinline__ float silu_f(float v) {
    return v * __builtin_amdgcn_rcpf(1.0f + __expf(-v));
}

__device__ __forceinline__ v8f vzero8() {
    v8f c;
#pragma unroll
    for (int i = 0; i < 8; ++i) c[i] = 0.0f;
    return c;
}

// A fragment (16x32 bf16) from an LDS tile laid out [16 rows][128 cols].
__device__ __forceinline__ v16bf load_a_frag(const __bf16* xs, int lane, int kbase) {
    int row = lane & 15;
    int hi  = (lane >> 4) << 3;  // 0 or 8
    const __bf16* p = xs + row * Hc + kbase;
    v16bf a;
#pragma unroll
    for (int e = 0; e < 8; ++e) a[e] = p[hi + e];
#pragma unroll
    for (int e = 0; e < 8; ++e) a[8 + e] = p[hi + 16 + e];
    return a;
}

// B fragment (32x16 bf16) from LDS weight [128 K][128 N].
__device__ __forceinline__ v16bf load_b_frag(const __bf16* wb, int lane, int kbase, int nbase) {
    int col = lane & 15;
    int hi  = (lane >> 4) << 3;  // 0 or 8
    const __bf16* p = wb + (kbase + hi) * Hc + nbase + col;
    v16bf b;
#pragma unroll
    for (int e = 0; e < 8; ++e) b[e] = p[e * Hc];
#pragma unroll
    for (int e = 0; e < 8; ++e) b[8 + e] = p[(16 + e) * Hc];
    return b;
}

// =====================================================================
// Weight pre-conversion (once per launch): f32 -> bf16, incl. W_bil
// transposed to [jb][j][i].
// =====================================================================
__global__ __launch_bounds__(256) void k_cvt_bf16(const float* __restrict__ src,
                                                  __bf16* __restrict__ dst, int n) {
    int i = blockIdx.x * 256 + threadIdx.x;
    if (i < n) dst[i] = (__bf16)src[i];
}

__global__ __launch_bounds__(256) void k_cvt_bil(const float* __restrict__ Wbil,
                                                 __bf16* __restrict__ dst) {
    int idx = blockIdx.x * 256 + threadIdx.x;     // over 8*128*128
    if (idx >= NBc * Hc * Hc) return;
    int jb = idx >> 14;
    int j  = (idx >> 7) & 127;
    int i_ = idx & 127;
    dst[idx] = (__bf16)Wbil[((size_t)i_ * NBc + jb) * Hc + j];
}

// =====================================================================
// Kernel 1: y[e,:] = silu(x[e,:] @ W + b) [* (rbf[e,:] @ W_rbf) if use_r]
// =====================================================================
__global__ __launch_bounds__(128) void k_edge_gemm(
    const float* __restrict__ x, const __bf16* __restrict__ Wb,
    const float* __restrict__ bias,
    const float* __restrict__ rbf, const float* __restrict__ W_rbf, int use_r,
    __bf16* __restrict__ yout, int E)
{
    __shared__ __align__(16) __bf16 wl[Hc * Hc];        // 32 KB
    __shared__ __align__(16) __bf16 xs[4][16 * Hc];     // 16 KB
    __shared__ float  bl[Hc];
    __shared__ float  wrbf[NRc * Hc];
    __shared__ float  rbf_s[4][16 * NRc];

    int tid = threadIdx.x, wave = tid >> 5, lane = tid & 31;
    int ebase = blockIdx.x * 64 + wave * 16;

    stage_w(wl, Wb, tid);                         // async -> LDS
    if (tid < Hc) bl[tid] = bias[tid];
    for (int k = tid; k < NRc * Hc; k += 128) wrbf[k] = W_rbf[k];

    for (int cc = lane; cc < 16 * (Hc / 4); cc += 32) {
        int r  = cc >> 5;
        int c4 = (cc & 31) * 4;
        int e = ebase + r; int es = e < E ? e : 0;
        float4 xv = ((const float4*)(x + (size_t)es * Hc))[c4 >> 2];
        __bf16* dp = &xs[wave][r * Hc + c4];
        dp[0] = (__bf16)xv.x; dp[1] = (__bf16)xv.y;
        dp[2] = (__bf16)xv.z; dp[3] = (__bf16)xv.w;
    }
    for (int k = lane; k < 16 * NRc; k += 32) {
        int r = k / NRc, c = k % NRc;
        int e = ebase + r; int es = e < E ? e : 0;
        rbf_s[wave][k] = rbf[(size_t)es * NRc + c];
    }
    async_wait();
    __syncthreads();

    v16bf a[4];
#pragma unroll
    for (int ks = 0; ks < 4; ++ks) a[ks] = load_a_frag(&xs[wave][0], lane, ks * 32);

    int colL = lane & 15, hi8 = (lane >> 4) << 3;
#pragma unroll
    for (int nt = 0; nt < 8; ++nt) {
        v8f c = vzero8();
#pragma unroll
        for (int ks = 0; ks < 4; ++ks) {
            v16bf b = load_b_frag(wl, lane, ks * 32, nt * 16);
            c = __builtin_amdgcn_wmma_f32_16x16x32_bf16(false, a[ks], false, b,
                                                        (short)0, c, false, false);
        }
        int ncol = nt * 16 + colL;
        float bv = bl[ncol];
#pragma unroll
        for (int v = 0; v < 8; ++v) {
            int row = v + hi8;
            int e = ebase + row;
            float val = silu_f(c[v] + bv);
            if (use_r) {
                float r6 = 0.0f;
#pragma unroll
                for (int q = 0; q < NRc; ++q)
                    r6 += rbf_s[wave][row * NRc + q] * wrbf[q * Hc + ncol];
                val *= r6;
            }
            if (e < E) yout[(size_t)e * Hc + ncol] = (__bf16)val;
        }
    }
}

// =====================================================================
// Kernel 2: sproj = sbf @ W_sbf
// =====================================================================
__global__ __launch_bounds__(256) void k_sproj(
    const float* __restrict__ sbf, const float* __restrict__ W_sbf,
    float* __restrict__ sproj, int T)
{
    __shared__ float ws[NSRc * NBc];
    for (int k = threadIdx.x; k < NSRc * NBc; k += 256) ws[k] = W_sbf[k];
    __syncthreads();
    int t = blockIdx.x * 256 + threadIdx.x;
    if (t >= T) return;
    float acc[NBc];
#pragma unroll
    for (int j = 0; j < NBc; ++j) acc[j] = 0.0f;
    const float* sr = sbf + (size_t)t * NSRc;
    for (int q = 0; q < NSRc; ++q) {
        float s = sr[q];
#pragma unroll
        for (int j = 0; j < NBc; ++j) acc[j] += s * ws[q * NBc + j];
    }
#pragma unroll
    for (int j = 0; j < NBc; ++j) sproj[(size_t)t * NBc + j] = acc[j];
}

// =====================================================================
// Kernel 3: triplet bilinear + scatter, W_bil slices double-buffered
// with async global->LDS copies overlapping the WMMA compute.
// =====================================================================
__global__ __launch_bounds__(128) void k_triplet(
    const __bf16* __restrict__ xkj, const float* __restrict__ sproj,
    const int* __restrict__ idx_kj, const int* __restrict__ idx_ji,
    const __bf16* __restrict__ Wbil_t,   // [jb][j][i] bf16
    float* __restrict__ agg, int T)
{
    __shared__ __align__(16) __bf16 wb[2][Hc * Hc];   // 64 KB double buffer
    __shared__ __align__(16) __bf16 xs[4][16 * Hc];   // 16 KB
    __shared__ float  sp[4][16 * NBc];
    __shared__ int    ij[4][16];

    int tid = threadIdx.x, wave = tid >> 5, lane = tid & 31;
    int tbase = blockIdx.x * 64 + wave * 16;

    // prologue: async copy of slice 0 while gathering the x_kj tile
    stage_w(wb[0], Wbil_t, tid);

    if (lane < 16) {
        int t = tbase + lane; int ts = t < T ? t : 0;
        ij[wave][lane] = idx_ji[ts];
    }
    for (int cc = lane; cc < 16 * (Hc / 8); cc += 32) {
        int r  = cc >> 4;
        int c8 = cc & 15;
        int t = tbase + r; int ts = t < T ? t : 0;
        int e = idx_kj[ts];
        uint4 v = ((const uint4*)(xkj + (size_t)e * Hc))[c8];
        ((uint4*)&xs[wave][r * Hc])[c8] = v;
    }
    for (int k = lane; k < 16 * NBc; k += 32) {
        int t = tbase + (k >> 3); int ts = t < T ? t : 0;
        sp[wave][k] = sproj[(size_t)ts * NBc + (k & 7)];
    }
    async_wait();
    __syncthreads();

    v16bf a[4];
#pragma unroll
    for (int ks = 0; ks < 4; ++ks) a[ks] = load_a_frag(&xs[wave][0], lane, ks * 32);

    v8f tot[8];
#pragma unroll
    for (int nt = 0; nt < 8; ++nt) tot[nt] = vzero8();

    int colL = lane & 15, hi8 = (lane >> 4) << 3;

    for (int jb = 0; jb < NBc; ++jb) {
        // issue async copy of next slice into the other buffer
        // (safe: barrier at end of previous iteration released that buffer)
        if (jb + 1 < NBc)
            stage_w(wb[(jb + 1) & 1], Wbil_t + (size_t)(jb + 1) * Hc * Hc, tid);

        const __bf16* cw = wb[jb & 1];
#pragma unroll
        for (int nt = 0; nt < 8; ++nt) {
            v8f c = vzero8();
#pragma unroll
            for (int ks = 0; ks < 4; ++ks) {
                v16bf b = load_b_frag(cw, lane, ks * 32, nt * 16);
                c = __builtin_amdgcn_wmma_f32_16x16x32_bf16(false, a[ks], false, b,
                                                            (short)0, c, false, false);
            }
#pragma unroll
            for (int v = 0; v < 8; ++v)
                tot[nt][v] += sp[wave][(v + hi8) * NBc + jb] * c[v];
        }
        async_wait();        // next slice landed
        __syncthreads();     // all waves done with current slice
    }

#pragma unroll
    for (int nt = 0; nt < 8; ++nt) {
        int ncol = nt * 16 + colL;
#pragma unroll
        for (int v = 0; v < 8; ++v) {
            int row = v + hi8;
            int t = tbase + row;
            if (t < T)
                atomicAdd(&agg[(size_t)ij[wave][row] * Hc + ncol], tot[nt][v]);
        }
    }
}

// =====================================================================
// Kernel 4: x2 = silu((x_ji + agg) @ W_lin + b_lin), in-place over x_ji
// =====================================================================
__global__ __launch_bounds__(128) void k_combine(
    const __bf16* xji, const float* __restrict__ agg,
    const __bf16* __restrict__ Wlinb, const float* __restrict__ b_lin,
    __bf16* x2, int E)
{
    __shared__ __align__(16) __bf16 wl[Hc * Hc];
    __shared__ __align__(16) __bf16 xs[4][16 * Hc];
    __shared__ float  bl[Hc];

    int tid = threadIdx.x, wave = tid >> 5, lane = tid & 31;
    int ebase = blockIdx.x * 64 + wave * 16;

    stage_w(wl, Wlinb, tid);
    if (tid < Hc) bl[tid] = b_lin[tid];
    for (int cc = lane; cc < 16 * (Hc / 4); cc += 32) {
        int r  = cc >> 5;
        int c4 = (cc & 31) * 4;
        int e = ebase + r; int es = e < E ? e : 0;
        size_t off = (size_t)es * Hc + c4;
        float4 av = *(const float4*)(agg + off);
        const __bf16* xp = xji + off;
        __bf16* dp = &xs[wave][r * Hc + c4];
        dp[0] = (__bf16)((float)xp[0] + av.x);
        dp[1] = (__bf16)((float)xp[1] + av.y);
        dp[2] = (__bf16)((float)xp[2] + av.z);
        dp[3] = (__bf16)((float)xp[3] + av.w);
    }
    async_wait();
    __syncthreads();

    v16bf a[4];
#pragma unroll
    for (int ks = 0; ks < 4; ++ks) a[ks] = load_a_frag(&xs[wave][0], lane, ks * 32);

    int colL = lane & 15, hi8 = (lane >> 4) << 3;
#pragma unroll
    for (int nt = 0; nt < 8; ++nt) {
        v8f c = vzero8();
#pragma unroll
        for (int ks = 0; ks < 4; ++ks) {
            v16bf b = load_b_frag(wl, lane, ks * 32, nt * 16);
            c = __builtin_amdgcn_wmma_f32_16x16x32_bf16(false, a[ks], false, b,
                                                        (short)0, c, false, false);
        }
        int ncol = nt * 16 + colL;
        float bv = bl[ncol];
#pragma unroll
        for (int v = 0; v < 8; ++v) {
            int e = ebase + v + hi8;
            if (e < E) x2[(size_t)e * Hc + ncol] = (__bf16)silu_f(c[v] + bv);
        }
    }
}

// =====================================================================
// Kernel 5a: output-block gate + scatter to nodes (4 edges per block).
// =====================================================================
#define EDPB 4
__global__ __launch_bounds__(128) void k_gate_scatter(
    const float* __restrict__ xe_f, const __bf16* __restrict__ xe_b,
    const float* __restrict__ rbf, const float* __restrict__ lin_rbf,
    const int* __restrict__ iidx, float* __restrict__ nodeg, int E)
{
    __shared__ float wl[NRc * Hc];
    __shared__ float rs[EDPB][NRc];
    int tid = threadIdx.x;
    int e0 = blockIdx.x * EDPB;
    for (int k = tid; k < NRc * Hc; k += 128) wl[k] = lin_rbf[k];
    if (tid < EDPB * NRc) {
        int ee = e0 + tid / NRc; int es = ee < E ? ee : 0;
        rs[tid / NRc][tid % NRc] = rbf[(size_t)es * NRc + (tid % NRc)];
    }
    __syncthreads();
    int c = tid;
    for (int j = 0; j < EDPB; ++j) {
        int e = e0 + j;
        if (e >= E) break;
        float p = 0.0f;
#pragma unroll
        for (int q = 0; q < NRc; ++q) p += rs[j][q] * wl[q * Hc + c];
        float xv = xe_f ? xe_f[(size_t)e * Hc + c] : (float)xe_b[(size_t)e * Hc + c];
        atomicAdd(&nodeg[(size_t)iidx[e] * Hc + c], p * xv);
    }
}

// =====================================================================
// Kernel 5b: node MLP (3 silu layers, WMMA) + final lin + column sum.
// =====================================================================
__global__ __launch_bounds__(128) void k_node_mlp(
    const float* __restrict__ nodeg, const __bf16* __restrict__ lins_Wb,
    const float* __restrict__ lins_b,
    const __bf16* __restrict__ lin_outb,
    float* __restrict__ outv, int N)
{
    __shared__ __align__(16) __bf16 wl[Hc * Hc];
    __shared__ __align__(16) __bf16 hs[4][16 * Hc];
    __shared__ float  bl[Hc];
    __shared__ float  colsum[Hc];

    int tid = threadIdx.x, wave = tid >> 5, lane = tid & 31;
    int nbase0 = blockIdx.x * 64 + wave * 16;
    int colL = lane & 15, hi8 = (lane >> 4) << 3;

    for (int cc = lane; cc < 16 * (Hc / 4); cc += 32) {
        int r  = cc >> 5;
        int c4 = (cc & 31) * 4;
        int n = nbase0 + r; int ns = n < N ? n : 0;
        float4 gv = ((const float4*)(nodeg + (size_t)ns * Hc))[c4 >> 2];
        if (n >= N) { gv.x = gv.y = gv.z = gv.w = 0.0f; }
        __bf16* dp = &hs[wave][r * Hc + c4];
        dp[0] = (__bf16)gv.x; dp[1] = (__bf16)gv.y;
        dp[2] = (__bf16)gv.z; dp[3] = (__bf16)gv.w;
    }
    if (tid < Hc) colsum[tid] = 0.0f;

    for (int layer = 0; layer < NLAYc; ++layer) {
        __syncthreads();
        stage_w(wl, lins_Wb + (size_t)layer * Hc * Hc, tid);
        if (tid < Hc) bl[tid] = lins_b[layer * Hc + tid];
        async_wait();
        __syncthreads();

        v16bf a[4];
#pragma unroll
        for (int ks = 0; ks < 4; ++ks) a[ks] = load_a_frag(&hs[wave][0], lane, ks * 32);

        float res[8][8];
#pragma unroll
        for (int nt = 0; nt < 8; ++nt) {
            v8f c = vzero8();
#pragma unroll
            for (int ks = 0; ks < 4; ++ks) {
                v16bf b = load_b_frag(wl, lane, ks * 32, nt * 16);
                c = __builtin_amdgcn_wmma_f32_16x16x32_bf16(false, a[ks], false, b,
                                                            (short)0, c, false, false);
            }
            float bv = bl[nt * 16 + colL];
#pragma unroll
            for (int v = 0; v < 8; ++v) res[nt][v] = silu_f(c[v] + bv);
        }
#pragma unroll
        for (int nt = 0; nt < 8; ++nt)
#pragma unroll
            for (int v = 0; v < 8; ++v)
                hs[wave][(v + hi8) * Hc + nt * 16 + colL] = (__bf16)res[nt][v];
    }

    __syncthreads();
    stage_w(wl, lin_outb, tid);
    async_wait();
    __syncthreads();

    v16bf a[4];
#pragma unroll
    for (int ks = 0; ks < 4; ++ks) a[ks] = load_a_frag(&hs[wave][0], lane, ks * 32);

#pragma unroll
    for (int nt = 0; nt < 8; ++nt) {
        v8f c = vzero8();
#pragma unroll
        for (int ks = 0; ks < 4; ++ks) {
            v16bf b = load_b_frag(wl, lane, ks * 32, nt * 16);
            c = __builtin_amdgcn_wmma_f32_16x16x32_bf16(false, a[ks], false, b,
                                                        (short)0, c, false, false);
        }
        float s = 0.0f;
#pragma unroll
        for (int v = 0; v < 8; ++v) {
            int n = nbase0 + v + hi8;
            if (n < N) s += c[v];
        }
        atomicAdd(&colsum[nt * 16 + colL], s);
    }
    __syncthreads();
    if (tid < Hc) atomicAdd(&outv[tid], colsum[tid]);
}

// =====================================================================
// Host launcher
// =====================================================================
extern "C" void kernel_launch(void* const* d_in, const int* in_sizes, int n_in,
                              void* d_out, int out_size, void* d_ws, size_t ws_size,
                              hipStream_t stream)
{
    const float* x       = (const float*)d_in[0];
    const float* rbf     = (const float*)d_in[1];
    const float* sbf     = (const float*)d_in[2];
    const int*   iidx    = (const int*)d_in[3];
    const int*   idx_kj  = (const int*)d_in[4];
    const int*   idx_ji  = (const int*)d_in[5];
    const float* W_rbf   = (const float*)d_in[6];
    const float* W_sbf   = (const float*)d_in[7];
    const float* W_kj    = (const float*)d_in[8];
    const float* b_kj    = (const float*)d_in[9];
    const float* W_ji    = (const float*)d_in[10];
    const float* b_ji    = (const float*)d_in[11];
    const float* W_bil   = (const float*)d_in[12];
    const float* W_lin   = (const float*)d_in[13];
    const float* b_lin   = (const float*)d_in[14];
    const float* o_lrbf  = (const float*)d_in[15];
    const float* o_linsW = (const float*)d_in[16];
    const float* o_linsb = (const float*)d_in[17];
    const float* o_lin   = (const float*)d_in[18];

    const int E = in_sizes[3];
    const int T = in_sizes[4];
    const int N = NNODES;
    const int HH = Hc * Hc;

    char* w = (char*)d_ws;
    float*  agg    = (float*)w;  w += (size_t)E * Hc * sizeof(float);
    __bf16* xkj    = (__bf16*)w; w += (size_t)E * Hc * sizeof(__bf16);
    __bf16* xji    = (__bf16*)w; w += (size_t)E * Hc * sizeof(__bf16);
    float*  sproj  = (float*)w;  w += (size_t)T * NBc * sizeof(float);
    float*  nodeg  = (float*)w;  w += (size_t)N * Hc * sizeof(float);
    __bf16* wb_ji   = (__bf16*)w; w += (size_t)HH * sizeof(__bf16);
    __bf16* wb_kj   = (__bf16*)w; w += (size_t)HH * sizeof(__bf16);
    __bf16* wb_lin  = (__bf16*)w; w += (size_t)HH * sizeof(__bf16);
    __bf16* wb_outW = (__bf16*)w; w += (size_t)2 * NLAYc * HH * sizeof(__bf16);
    __bf16* wb_outL = (__bf16*)w; w += (size_t)2 * HH * sizeof(__bf16);
    __bf16* wb_bilt = (__bf16*)w; w += (size_t)NBc * HH * sizeof(__bf16);
    float*  outp    = (float*)d_out;

    (void)hipMemsetAsync(outp, 0, OUTc * sizeof(float), stream);
    (void)hipMemsetAsync(agg, 0, (size_t)E * Hc * sizeof(float), stream);
    (void)hipMemsetAsync(nodeg, 0, (size_t)N * Hc * sizeof(float), stream);

    int cg = (HH + 255) / 256;
    k_cvt_bf16<<<cg, 256, 0, stream>>>(W_ji, wb_ji, HH);
    k_cvt_bf16<<<cg, 256, 0, stream>>>(W_kj, wb_kj, HH);
    k_cvt_bf16<<<cg, 256, 0, stream>>>(W_lin, wb_lin, HH);
    k_cvt_bf16<<<(2 * NLAYc * HH + 255) / 256, 256, 0, stream>>>(o_linsW, wb_outW, 2 * NLAYc * HH);
    k_cvt_bf16<<<(2 * HH + 255) / 256, 256, 0, stream>>>(o_lin, wb_outL, 2 * HH);
    k_cvt_bil<<<(NBc * HH + 255) / 256, 256, 0, stream>>>(W_bil, wb_bilt);

    int egrid = (E + 63) / 64;
    int ngrid = (N + 63) / 64;
    int ggrid = (E + EDPB - 1) / EDPB;

    // ---- output block 0 (on input x) ----
    k_gate_scatter<<<ggrid, 128, 0, stream>>>(x, (const __bf16*)nullptr, rbf,
                                              o_lrbf + 0, iidx, nodeg, E);
    k_node_mlp<<<ngrid, 128, 0, stream>>>(nodeg, wb_outW + 0, o_linsb + 0,
                                          wb_outL + 0, outp, N);

    // ---- interaction block ----
    k_edge_gemm<<<egrid, 128, 0, stream>>>(x, wb_ji, b_ji, rbf, W_rbf, 0, xji, E);
    k_edge_gemm<<<egrid, 128, 0, stream>>>(x, wb_kj, b_kj, rbf, W_rbf, 1, xkj, E);
    k_sproj<<<(T + 255) / 256, 256, 0, stream>>>(sbf, W_sbf, sproj, T);
    k_triplet<<<(T + 63) / 64, 128, 0, stream>>>(xkj, sproj, idx_kj, idx_ji,
                                                 wb_bilt, agg, T);
    k_combine<<<egrid, 128, 0, stream>>>(xji, agg, wb_lin, b_lin, xji, E);

    // ---- output block 1 (on x2, stored bf16 in xji buffer) ----
    (void)hipMemsetAsync(nodeg, 0, (size_t)N * Hc * sizeof(float), stream);
    k_gate_scatter<<<ggrid, 128, 0, stream>>>((const float*)nullptr, xji, rbf,
                                              o_lrbf + NRc * Hc, iidx, nodeg, E);
    k_node_mlp<<<ngrid, 128, 0, stream>>>(nodeg, wb_outW + (size_t)NLAYc * HH,
                                          o_linsb + NLAYc * Hc,
                                          wb_outL + HH, outp, N);
}
